// PI05Model_29317446763239
// MI455X (gfx1250) — matrix operand order repaired and checked by reading, not compiled, not run on hardware
//
#include <hip/hip_runtime.h>
#include <math.h>

// ---------------------------------------------------------------------------
// Types / helpers
// ---------------------------------------------------------------------------
typedef __attribute__((ext_vector_type(16))) __bf16 bf16x16;
typedef __attribute__((ext_vector_type(8)))  float  f32x8;

__device__ __forceinline__ __bf16 tobf(float f) {
  unsigned u = __builtin_bit_cast(unsigned, f);
  unsigned r = u + 0x7FFFu + ((u >> 16) & 1u);        // round-to-nearest-even
  unsigned short h = (unsigned short)(r >> 16);
  return __builtin_bit_cast(__bf16, h);
}
__device__ __forceinline__ unsigned short bfbits(__bf16 b) {
  return __builtin_bit_cast(unsigned short, b);
}
__device__ __forceinline__ float gelu_tanh(float x) {
  float t = 0.7978845608028654f * (x + 0.044715f * x * x * x);
  return 0.5f * x * (1.0f + tanhf(t));
}

// Load 8 consecutive elements of B (fp32 weights or bf16 acts) as bf16
template <typename TB>
__device__ __forceinline__ void loadrow8(const TB* p, __bf16 o[8]) {
  if constexpr (sizeof(TB) == 4) {
    float4 a = *(const float4*)p;
    float4 b = *(const float4*)(p + 4);
    o[0] = tobf(a.x); o[1] = tobf(a.y); o[2] = tobf(a.z); o[3] = tobf(a.w);
    o[4] = tobf(b.x); o[5] = tobf(b.y); o[6] = tobf(b.z); o[7] = tobf(b.w);
  } else {
    uint4 u = *(const uint4*)p;
    unsigned w[4] = {u.x, u.y, u.z, u.w};
    #pragma unroll
    for (int j = 0; j < 4; j++) {
      o[2 * j]     = __builtin_bit_cast(__bf16, (unsigned short)(w[j] & 0xFFFFu));
      o[2 * j + 1] = __builtin_bit_cast(__bf16, (unsigned short)(w[j] >> 16));
    }
  }
}

// ---------------------------------------------------------------------------
// Generic WMMA GEMM: C = epilogue(A[MxK](bf16) * B[KxN]), fp32 accumulate.
//   TB      : element type of B in global memory (float weights or __bf16)
//   BTRANS  : B stored as [N][K] (scores: B = k of shape [Skeys][Dh]); may
//             have ragged N.  Non-BTRANS callers guarantee N % BN == 0.
//   EPI     : compile-time epilogue selector.  EPI_DUAL = two B matrices,
//             epilogue gelu(accG)*accU -> bf16 (gated MLP); uses a 128x64
//             block tile to halve acc registers, thread halves stage B0/B1.
// 256 threads (8 wave32), wave tile 32 x (16*NF), double-buffered LDS.
// ---------------------------------------------------------------------------
struct ZMap { int div; long outer; long inner; };
struct GemmParams {
  const void* A; int lda; int aRemap, aRowsPerB, aBatchRows, aRowOff; ZMap aZ;
  const void* B0; const void* B1; int ldb; ZMap bZ;
  void* C; int ldc; int cRemap, cRowsPerB, cBatchRows, cRowOff; ZMap cZ;
  const float* res;  int ldres;
  const float* gate; int gateLd; int gateRowsPerB;  // g = gate[(r/gateRowsPerB)*gateLd + col]
  const float* mask; int ldmask; ZMap mZ;
  float scale;
  int M, N, K;
};
enum { EPI_F32 = 0, EPI_BF16 = 1, EPI_SCALE_MASK = 2, EPI_RES = 3, EPI_DUAL = 4 };

__device__ __forceinline__ long zoff(ZMap m, int z) {
  return (long)(z / m.div) * m.outer + (long)(z % m.div) * m.inner;
}

template <typename TB, bool BTRANS, int EPI>
__global__ __launch_bounds__(256) void gemm_wmma(GemmParams p) {
  constexpr bool DUAL = (EPI == EPI_DUAL);
  constexpr int NF = DUAL ? 2 : 4;       // B fragments per wave
  constexpr int BM = 128, BN = 32 * NF, BK = 32, PAD = 40;
  __shared__ __bf16 Alds[2][BM][PAD];
  __shared__ __bf16 Blds0[2][BN][PAD];
  __shared__ __bf16 Blds1[DUAL ? 2 : 1][BN][PAD];

  const int tid  = threadIdx.x;
  const int lane = tid & 31, wid = tid >> 5;
  const int wm = wid & 3, wn = wid >> 2;          // wave position in 4x2 grid
  const int hlf = lane >> 4, l16 = lane & 15;
  const int z = blockIdx.z;

  const __bf16* A  = (const __bf16*)p.A + zoff(p.aZ, z);
  const TB*     B0 = (const TB*)p.B0 + zoff(p.bZ, z);
  const TB*     B1 = DUAL ? ((const TB*)p.B1 + zoff(p.bZ, z)) : nullptr;

  const int row0 = blockIdx.y * BM;
  const int col0 = blockIdx.x * BN;

  f32x8 zero8 = {0.f, 0.f, 0.f, 0.f, 0.f, 0.f, 0.f, 0.f};
  f32x8 acc0[2][NF], acc1[2][NF];
  for (int i = 0; i < 2; i++)
    for (int j = 0; j < NF; j++) { acc0[i][j] = zero8; acc1[i][j] = zero8; }

  // A staging: 2 threads per row, 16 bf16 (32B) each
  const int sr = tid >> 1;
  const int sc = (tid & 1) << 4;

  // A-side row remap is loop invariant
  long agr = 0; const bool arowOk = (row0 + sr) < p.M;
  if (arowOk) {
    int r = row0 + sr;
    agr = p.aRemap
        ? (long)(r / p.aRowsPerB) * p.aBatchRows + (r % p.aRowsPerB) + p.aRowOff
        : (long)r;
  }

  auto stage = [&](int buf, int k0) {
    // ---- A tile ----
    {
      uint4 u0 = {0, 0, 0, 0}, u1 = {0, 0, 0, 0};
      if (arowOk) {
        const __bf16* ap = A + agr * (long)p.lda + k0 + sc;
        const uint4* src = (const uint4*)ap;
        u0 = src[0]; u1 = src[1];
        if (k0 + BK < p.K) __builtin_prefetch(ap + BK, 0, 1);
      }
      *(uint4*)&Alds[buf][sr][sc]     = u0;
      *(uint4*)&Alds[buf][sr][sc + 8] = u1;
    }
    // ---- B tile as Blds[n][k] ----
    if constexpr (BTRANS) {
      uint4 u0 = {0, 0, 0, 0}, u1 = {0, 0, 0, 0};
      int n = col0 + sr;
      if (n < p.N) {
        const uint4* src = (const uint4*)(B0 + (long)n * p.ldb + k0 + sc);
        u0 = src[0]; u1 = src[1];
      }
      *(uint4*)&Blds0[buf][sr][sc]     = u0;
      *(uint4*)&Blds0[buf][sr][sc + 8] = u1;
    } else if constexpr (DUAL) {
      // BN=64: threads 0..127 stage B0, threads 128..255 stage B1
      const int t = tid & 127;
      const int kp2 = (t & 15) << 1;       // even k offset 0..30
      const int nb8 = ((t >> 4) & 7) << 3; // column offset 0..56 step 8
      const TB* Bsel = (tid < 128) ? B0 : B1;
      __bf16* lds = (tid < 128) ? &Blds0[buf][0][0] : &Blds1[buf][0][0];
      int kk = k0 + kp2;
      const TB* br = Bsel + (long)kk * p.ldb + col0 + nb8;
      __bf16 r0[8], r1[8];
      loadrow8(br, r0);
      loadrow8(br + p.ldb, r1);
      #pragma unroll
      for (int j = 0; j < 8; j++) {
        unsigned pk = (unsigned)bfbits(r0[j]) | ((unsigned)bfbits(r1[j]) << 16);
        *(unsigned*)&lds[(long)(nb8 + j) * PAD + kp2] = pk;
      }
    } else {
      // BN=128, N % 128 == 0 guaranteed by callers
      const int kp2 = (tid & 15) << 1;   // even k offset 0..30
      const int nb8 = (tid >> 4) << 3;   // column offset 0..120 step 8
      int kk = k0 + kp2;
      const TB* br = B0 + (long)kk * p.ldb + col0 + nb8;
      __bf16 r0[8], r1[8];
      loadrow8(br, r0);
      loadrow8(br + p.ldb, r1);
      #pragma unroll
      for (int j = 0; j < 8; j++) {
        unsigned pk = (unsigned)bfbits(r0[j]) | ((unsigned)bfbits(r1[j]) << 16);
        *(unsigned*)&Blds0[buf][nb8 + j][kp2] = pk;
      }
    }
  };

  stage(0, 0);
  __syncthreads();

  int buf = 0;
  for (int k0 = 0; k0 < p.K; k0 += BK) {
    if (k0 + BK < p.K) stage(buf ^ 1, k0 + BK);

    // ---- fragments: contiguous k = 16*hlf + s, same order for A and B ----
    union FragB { bf16x16 v; uint4 u[2]; };
    FragB fa[2], fb[NF], fb1[NF];
    #pragma unroll
    for (int mi = 0; mi < 2; mi++) {
      const __bf16* base = &Alds[buf][wm * 32 + mi * 16 + l16][0];
      fa[mi].u[0] = *(const uint4*)(base + hlf * 16);       // K = 16*hlf .. +7
      fa[mi].u[1] = *(const uint4*)(base + hlf * 16 + 8);   // K = 16*hlf+8 .. +15
    }
    #pragma unroll
    for (int ni = 0; ni < NF; ni++) {
      const __bf16* base = &Blds0[buf][wn * 16 * NF + ni * 16 + l16][0];
      fb[ni].u[0] = *(const uint4*)(base + hlf * 16);       // K = 16*hlf .. +15
      fb[ni].u[1] = *(const uint4*)(base + hlf * 16 + 8);
      if constexpr (DUAL) {
        const __bf16* b1 = &Blds1[buf][wn * 16 * NF + ni * 16 + l16][0];
        fb1[ni].u[0] = *(const uint4*)(b1 + hlf * 16);
        fb1[ni].u[1] = *(const uint4*)(b1 + hlf * 16 + 8);
      }
    }
    #pragma unroll
    for (int mi = 0; mi < 2; mi++)
      #pragma unroll
      for (int ni = 0; ni < NF; ni++) {
        acc0[mi][ni] = __builtin_amdgcn_wmma_f32_16x16x32_bf16(
            false, fa[mi].v, false, fb[ni].v, (short)0, acc0[mi][ni], false, false);
        if constexpr (DUAL)
          acc1[mi][ni] = __builtin_amdgcn_wmma_f32_16x16x32_bf16(
              false, fa[mi].v, false, fb1[ni].v, (short)0, acc1[mi][ni], false, false);
      }
    __syncthreads();
    buf ^= 1;
  }

  // ---- epilogue: C element (VGPR v) is M = v + 8*hlf, N = l16 ----
  // Fully specialized at compile time; row remap hoisted out of column loop;
  // identity maps do no integer division.
  const long czo = zoff(p.cZ, z);
  for (int mi = 0; mi < 2; mi++)
    #pragma unroll
    for (int v = 0; v < 8; v++) {
      int r = row0 + wm * 32 + mi * 16 + v + 8 * hlf;
      if (r >= p.M) continue;
      long orow = p.cRemap
          ? (long)(r / p.cRowsPerB) * p.cBatchRows + (r % p.cRowsPerB) + p.cRowOff
          : (long)r;
      long obase = czo + orow * (long)p.ldc;
      const float* grow = nullptr;
      if constexpr (EPI == EPI_RES)
        if (p.gate) grow = p.gate + (long)(r / p.gateRowsPerB) * p.gateLd;
      for (int ni = 0; ni < NF; ni++) {
        int c = col0 + wn * 16 * NF + ni * 16 + l16;
        if (c >= p.N) continue;
        float val = acc0[mi][ni][v];
        if constexpr (EPI == EPI_DUAL) {
          float u = acc1[mi][ni][v];
          ((__bf16*)p.C)[obase + c] = tobf(gelu_tanh(val) * u);
        } else if constexpr (EPI == EPI_F32) {
          ((float*)p.C)[obase + c] = val;
        } else if constexpr (EPI == EPI_BF16) {
          ((__bf16*)p.C)[obase + c] = tobf(val);
        } else if constexpr (EPI == EPI_SCALE_MASK) {
          float m = p.mask[zoff(p.mZ, z) + (long)r * p.ldmask + c];
          ((float*)p.C)[obase + c] = val * p.scale + m;
        } else { // EPI_RES: out = res + gate * acc
          float g = grow ? grow[c] : 1.0f;
          ((float*)p.C)[obase + c] = p.res[(long)r * p.ldres + c] + g * val;
        }
      }
    }
}

// ---------------------------------------------------------------------------
// Small kernels
// ---------------------------------------------------------------------------

// mod = adarms_cond @ W + b   (M=4 -> plain dot products)
__global__ __launch_bounds__(256) void mod_gemm_kernel(
    const float* __restrict__ a, const float* __restrict__ w,
    const float* __restrict__ bias, float* __restrict__ out,
    int K, int N, int B) {
  int idx = blockIdx.x * 256 + threadIdx.x;
  int n = idx % N, b = idx / N;
  if (b >= B) return;
  float acc = bias[n];
  for (int k = 0; k < K; k++) acc += a[(long)b * K + k] * w[(long)k * N + n];
  out[(long)b * N + n] = acc;
}

// RMSNorm; if mod != null: y = rms(x)*(1+sc)+sh with sc=mod[b][c], sh=mod[b][shOff+c]
// else:                    y = rms(x)*(1+ln[c]).  Output bf16.
__global__ __launch_bounds__(256) void rms_kernel(
    const float* __restrict__ x, __bf16* __restrict__ out, int D,
    const float* __restrict__ ln, const float* __restrict__ mod,
    int modLd, int shOff, int rowsPerB) {
  __shared__ float red[8];
  __shared__ float bc;
  int row = blockIdx.x;
  const float* xr = x + (long)row * D;
  float ss = 0.f;
  for (int c = threadIdx.x; c < D; c += 256) { float v = xr[c]; ss += v * v; }
  for (int o = 16; o > 0; o >>= 1) ss += __shfl_down(ss, o, 32);
  if ((threadIdx.x & 31) == 0) red[threadIdx.x >> 5] = ss;
  __syncthreads();
  if (threadIdx.x == 0) {
    float t = 0.f;
    for (int i = 0; i < 8; i++) t += red[i];
    bc = rsqrtf(t / (float)D + 1e-6f);
  }
  __syncthreads();
  float inv = bc;
  int b = row / rowsPerB;
  for (int c = threadIdx.x; c < D; c += 256) {
    float v = xr[c] * inv;
    if (mod) {
      float scv = mod[(long)b * modLd + c];
      float shv = mod[(long)b * modLd + shOff + c];
      v = v * (1.f + scv) + shv;
    } else {
      v = v * (1.f + ln[c]);
    }
    out[(long)row * D + c] = tobf(v);
  }
}

// RoPE for q: gathers prefix/expert, transposes [B,S,H,D]->[B,H,S,D], -> bf16
__global__ __launch_bounds__(256) void rope_q_kernel(
    const float* __restrict__ qp, const float* __restrict__ qe,
    const int* __restrict__ pos, __bf16* __restrict__ q) {
  const long total = (long)4 * 8 * 832 * 128;
  long idx = (long)blockIdx.x * 256 + threadIdx.x;
  if (idx >= total) return;
  int i = (int)(idx & 127);
  long t = idx >> 7;
  int s = (int)(t % 832); t /= 832;
  int h = (int)(t % 8);
  int b = (int)(t / 8);
  float ang = (float)pos[b * 832 + s] * __expf(-(float)i * (9.210340371976184f / 128.0f));
  float cs = __cosf(ang), sn = __sinf(ang);
  const float* src = (s < 768)
      ? (qp + ((long)(b * 768 + s) * 2048 + h * 256))
      : (qe + ((long)(b * 64 + (s - 768)) * 2048 + h * 256));
  float x1 = src[i], x2 = src[i + 128];
  __bf16* dst = q + (((long)(b * 8 + h) * 832 + s) * 256);
  dst[i]       = tobf(x1 * cs - x2 * sn);
  dst[i + 128] = tobf(x2 * cs + x1 * sn);
}

// RoPE for k: concat prefix/expert -> [B,832,256] bf16
__global__ __launch_bounds__(256) void rope_k_kernel(
    const float* __restrict__ kp, const float* __restrict__ ke,
    const int* __restrict__ pos, __bf16* __restrict__ k) {
  const long total = (long)4 * 832 * 128;
  long idx = (long)blockIdx.x * 256 + threadIdx.x;
  if (idx >= total) return;
  int i = (int)(idx & 127);
  long t = idx >> 7;
  int s = (int)(t % 832);
  int b = (int)(t / 832);
  float ang = (float)pos[b * 832 + s] * __expf(-(float)i * (9.210340371976184f / 128.0f));
  float cs = __cosf(ang), sn = __sinf(ang);
  const float* src = (s < 768) ? (kp + (long)(b * 768 + s) * 256)
                               : (ke + (long)(b * 64 + (s - 768)) * 256);
  float x1 = src[i], x2 = src[i + 128];
  __bf16* dst = k + ((long)(b * 832 + s) * 256);
  dst[i]       = tobf(x1 * cs - x2 * sn);
  dst[i + 128] = tobf(x2 * cs + x1 * sn);
}

// Row softmax (f32 in, bf16 out), one block per row
__global__ __launch_bounds__(256) void softmax_kernel(
    const float* __restrict__ s, __bf16* __restrict__ p, int S) {
  __shared__ float red[8];
  __shared__ float bc0, bc1;
  long row = blockIdx.x;
  const float* sr = s + row * (long)S;
  float mx = -3.0e38f;
  for (int c = threadIdx.x; c < S; c += 256) mx = fmaxf(mx, sr[c]);
  for (int o = 16; o > 0; o >>= 1) mx = fmaxf(mx, __shfl_down(mx, o, 32));
  if ((threadIdx.x & 31) == 0) red[threadIdx.x >> 5] = mx;
  __syncthreads();
  if (threadIdx.x == 0) {
    float t = red[0];
    for (int i = 1; i < 8; i++) t = fmaxf(t, red[i]);
    bc0 = t;
  }
  __syncthreads();
  mx = bc0;
  float sum = 0.f;
  for (int c = threadIdx.x; c < S; c += 256) sum += __expf(sr[c] - mx);
  for (int o = 16; o > 0; o >>= 1) sum += __shfl_down(sum, o, 32);
  if ((threadIdx.x & 31) == 0) red[threadIdx.x >> 5] = sum;
  __syncthreads();
  if (threadIdx.x == 0) {
    float t = 0.f;
    for (int i = 0; i < 8; i++) t += red[i];
    bc1 = 1.0f / t;
  }
  __syncthreads();
  float inv = bc1;
  for (int c = threadIdx.x; c < S; c += 256)
    p[row * (long)S + c] = tobf(__expf(sr[c] - mx) * inv);
}

// ---------------------------------------------------------------------------
// Host orchestration
// ---------------------------------------------------------------------------
static GemmParams mkParams(const void* A, int lda, const void* B0, const void* B1,
                           int ldb, void* C, int ldc, int M, int N, int K) {
  GemmParams p{};
  p.A = A; p.lda = lda; p.aRemap = 0; p.aRowsPerB = M; p.aBatchRows = 0; p.aRowOff = 0;
  p.aZ = {1, 0, 0};
  p.B0 = B0; p.B1 = B1; p.ldb = ldb; p.bZ = {1, 0, 0};
  p.C = C; p.ldc = ldc; p.cRemap = 0; p.cRowsPerB = M; p.cBatchRows = 0; p.cRowOff = 0;
  p.cZ = {1, 0, 0};
  p.res = nullptr; p.ldres = 0; p.gate = nullptr; p.gateLd = 0; p.gateRowsPerB = 1;
  p.mask = nullptr; p.ldmask = 0; p.mZ = {1, 0, 0};
  p.scale = 1.0f; p.M = M; p.N = N; p.K = K;
  return p;
}

#define LAUNCH_GEMM(TB, BT, EPI, prm, Z, TN)                              \
  do {                                                                    \
    dim3 _g(((prm).N + (TN) - 1) / (TN), ((prm).M + 127) / 128, (Z));     \
    gemm_wmma<TB, BT, EPI><<<_g, 256, 0, stream>>>(prm);                  \
  } while (0)

extern "C" void kernel_launch(void* const* d_in, const int* in_sizes, int n_in,
                              void* d_out, int out_size, void* d_ws, size_t ws_size,
                              hipStream_t stream) {
  (void)in_sizes; (void)n_in; (void)out_size; (void)ws_size;
  const float* x_prefix  = (const float*)d_in[0];
  const float* x_expert  = (const float*)d_in[1];
  const float* attn_mask = (const float*)d_in[2];
  const float* adarms    = (const float*)d_in[3];
  const int*   pos       = (const int*)d_in[4];
  const float* p_ln1     = (const float*)d_in[5];
  const float* p_wq      = (const float*)d_in[6];
  const float* p_wk      = (const float*)d_in[7];
  const float* p_wv      = (const float*)d_in[8];
  const float* p_wo      = (const float*)d_in[9];
  const float* p_ln2     = (const float*)d_in[10];
  const float* p_wgate   = (const float*)d_in[11];
  const float* p_wup     = (const float*)d_in[12];
  const float* p_wdown   = (const float*)d_in[13];
  const float* e_ada1_w  = (const float*)d_in[14];
  const float* e_ada1_b  = (const float*)d_in[15];
  const float* e_wq      = (const float*)d_in[16];
  const float* e_wk      = (const float*)d_in[17];
  const float* e_wv      = (const float*)d_in[18];
  const float* e_wo      = (const float*)d_in[19];
  const float* e_ada2_w  = (const float*)d_in[20];
  const float* e_ada2_b  = (const float*)d_in[21];
  const float* e_wgate   = (const float*)d_in[22];
  const float* e_wup     = (const float*)d_in[23];
  const float* e_wdown   = (const float*)d_in[24];

  constexpr int Bn = 4, S1 = 768, S2 = 64, S = 832;
  constexpr int Dp = 2048, De = 1024, Dh = 256, H = 8, Fp = 16384, Fe = 4096, HD = 2048;
  const int MP = Bn * S1;   // 3072 prefix rows
  const int ME = Bn * S2;   // 256 expert rows

  // ---- workspace bump allocator ----
  char* base = (char*)d_ws;
  size_t off = 0;
  auto alloc = [&](size_t bytes) -> char* {
    off = (off + 255) & ~(size_t)255;
    char* pp = base + off;
    off += bytes;
    return pp;
  };
  __bf16* h_p   = (__bf16*)alloc((size_t)MP * Dp * 2);
  __bf16* h_e   = (__bf16*)alloc((size_t)ME * De * 2);
  float*  mod1  = (float*) alloc((size_t)Bn * 3 * De * 4);
  float*  mod2  = (float*) alloc((size_t)Bn * 3 * De * 4);
  float*  qp    = (float*) alloc((size_t)MP * HD * 4);
  float*  qe    = (float*) alloc((size_t)ME * HD * 4);
  float*  kp    = (float*) alloc((size_t)MP * Dh * 4);
  float*  ke    = (float*) alloc((size_t)ME * Dh * 4);
  __bf16* qb    = (__bf16*)alloc((size_t)Bn * H * S * Dh * 2);
  __bf16* kb    = (__bf16*)alloc((size_t)Bn * S * Dh * 2);
  __bf16* vb    = (__bf16*)alloc((size_t)Bn * S * Dh * 2);
  float*  scores= (float*) alloc((size_t)Bn * H * S * S * 4);
  __bf16* probs = (__bf16*)alloc((size_t)Bn * H * S * S * 2);
  __bf16* att   = (__bf16*)alloc((size_t)Bn * S * HD * 2);
  float*  res_p = (float*) alloc((size_t)MP * Dp * 4);
  float*  res_e = (float*) alloc((size_t)ME * De * 4);
  __bf16* h2p   = (__bf16*)alloc((size_t)MP * Dp * 2);
  __bf16* h2e   = (__bf16*)alloc((size_t)ME * De * 2);
  __bf16* act_e = (__bf16*)alloc((size_t)ME * Fe * 2);
  // act_p (3072x16384 bf16 = 100MB) aliases dead scores+probs region (133MB)
  __bf16* act_p = (__bf16*)scores;

  // ---- adaLN modulation ----
  {
    int total = Bn * 3 * De;
    mod_gemm_kernel<<<(total + 255) / 256, 256, 0, stream>>>(adarms, e_ada1_w, e_ada1_b, mod1, De, 3 * De, Bn);
    mod_gemm_kernel<<<(total + 255) / 256, 256, 0, stream>>>(adarms, e_ada2_w, e_ada2_b, mod2, De, 3 * De, Bn);
  }

  // ---- pre-attention norms ----
  rms_kernel<<<MP, 256, 0, stream>>>(x_prefix, h_p, Dp, p_ln1, nullptr, 0, 0, MP);
  rms_kernel<<<ME, 256, 0, stream>>>(x_expert, h_e, De, nullptr, mod1, 3 * De, De, S2);

  // ---- QKV projections ----
  { GemmParams p = mkParams(h_p, Dp, p_wq, nullptr, HD, qp, HD, MP, HD, Dp);
    LAUNCH_GEMM(float, false, EPI_F32, p, 1, 128); }
  { GemmParams p = mkParams(h_p, Dp, p_wk, nullptr, Dh, kp, Dh, MP, Dh, Dp);
    LAUNCH_GEMM(float, false, EPI_F32, p, 1, 128); }
  { GemmParams p = mkParams(h_p, Dp, p_wv, nullptr, Dh, vb, Dh, MP, Dh, Dp);
    p.cRemap = 1; p.cRowsPerB = S1; p.cBatchRows = S; p.cRowOff = 0;  // pack v_p
    LAUNCH_GEMM(float, false, EPI_BF16, p, 1, 128); }
  { GemmParams p = mkParams(h_e, De, e_wq, nullptr, HD, qe, HD, ME, HD, De);
    LAUNCH_GEMM(float, false, EPI_F32, p, 1, 128); }
  { GemmParams p = mkParams(h_e, De, e_wk, nullptr, Dh, ke, Dh, ME, Dh, De);
    LAUNCH_GEMM(float, false, EPI_F32, p, 1, 128); }
  { GemmParams p = mkParams(h_e, De, e_wv, nullptr, Dh, vb, Dh, ME, Dh, De);
    p.cRemap = 1; p.cRowsPerB = S2; p.cBatchRows = S; p.cRowOff = S1; // pack v_e
    LAUNCH_GEMM(float, false, EPI_BF16, p, 1, 128); }

  // ---- RoPE (+ head transpose + concat + bf16) ----
  { long tq = (long)Bn * H * S * (Dh / 2);
    rope_q_kernel<<<(int)((tq + 255) / 256), 256, 0, stream>>>(qp, qe, pos, qb);
    long tk = (long)Bn * S * (Dh / 2);
    rope_k_kernel<<<(int)((tk + 255) / 256), 256, 0, stream>>>(kp, ke, pos, kb); }

  // ---- scores = q @ k^T * scale + mask   (Z = B*H batched) ----
  { GemmParams p = mkParams(qb, Dh, kb, nullptr, Dh, scores, S, S, S, Dh);
    p.aZ = {1, (long)S * Dh, 0};
    p.bZ = {H, (long)S * Dh, 0};
    p.cZ = {1, (long)S * S, 0};
    p.mask = attn_mask; p.ldmask = S; p.mZ = {H, (long)S * S, 0};
    p.scale = 0.0625f;    // 256^-0.5
    LAUNCH_GEMM(__bf16, true, EPI_SCALE_MASK, p, Bn * H, 128); }

  softmax_kernel<<<Bn * H * S, 256, 0, stream>>>(scores, probs, S);

  // ---- att = probs @ v, written as [B,S,H*Dh] bf16 ----
  { GemmParams p = mkParams(probs, S, vb, nullptr, Dh, att, HD, S, Dh, S);
    p.aZ = {1, (long)S * S, 0};
    p.bZ = {H, (long)S * Dh, 0};
    p.cZ = {H, (long)S * HD, (long)Dh};
    LAUNCH_GEMM(__bf16, false, EPI_BF16, p, Bn * H, 128); }

  // ---- output projections + residual ----
  { GemmParams p = mkParams(att, HD, p_wo, nullptr, Dp, res_p, Dp, MP, Dp, HD);
    p.aRemap = 1; p.aRowsPerB = S1; p.aBatchRows = S; p.aRowOff = 0;
    p.res = x_prefix; p.ldres = Dp;
    LAUNCH_GEMM(float, false, EPI_RES, p, 1, 128); }
  { GemmParams p = mkParams(att, HD, e_wo, nullptr, De, res_e, De, ME, De, HD);
    p.aRemap = 1; p.aRowsPerB = S2; p.aBatchRows = S; p.aRowOff = S1;
    p.res = x_expert; p.ldres = De;
    p.gate = mod1 + 2 * De; p.gateLd = 3 * De; p.gateRowsPerB = S2;
    LAUNCH_GEMM(float, false, EPI_RES, p, 1, 128); }

  // ---- post-attention norms ----
  rms_kernel<<<MP, 256, 0, stream>>>(res_p, h2p, Dp, p_ln2, nullptr, 0, 0, MP);
  rms_kernel<<<ME, 256, 0, stream>>>(res_e, h2e, De, nullptr, mod2, 3 * De, De, S2);

  // ---- gated MLPs: act = gelu(h@Wg) * (h@Wu) fused dual-GEMM (tile 128x64) ----
  { GemmParams p = mkParams(h2p, Dp, p_wgate, p_wup, Fp, act_p, Fp, MP, Fp, Dp);
    LAUNCH_GEMM(float, false, EPI_DUAL, p, 1, 64); }
  { GemmParams p = mkParams(h2e, De, e_wgate, e_wup, Fe, act_e, Fe, ME, Fe, De);
    LAUNCH_GEMM(float, false, EPI_DUAL, p, 1, 64); }

  // ---- down projections + residual (+gate) -> d_out ----
  float* out_p = (float*)d_out;
  float* out_e = out_p + (size_t)MP * Dp;
  { GemmParams p = mkParams(act_p, Fp, p_wdown, nullptr, Dp, out_p, Dp, MP, Dp, Fp);
    p.res = res_p; p.ldres = Dp;
    LAUNCH_GEMM(float, false, EPI_RES, p, 1, 128); }
  { GemmParams p = mkParams(act_e, Fe, e_wdown, nullptr, De, out_e, De, ME, De, Fe);
    p.res = res_e; p.ldres = De;
    p.gate = mod2 + 2 * De; p.gateLd = 3 * De; p.gateRowsPerB = S2;
    LAUNCH_GEMM(float, false, EPI_RES, p, 1, 128); }
}
